// SimpleRNNModel_60979945669251
// MI455X (gfx1250) — matrix operand order repaired and checked by reading, not compiled, not run on hardware
//
#include <hip/hip_runtime.h>
#include <hip/hip_bf16.h>

// ---------------------------------------------------------------------------
// SimpleRNN (2-layer tanh RNN) for MI455X / gfx1250, bf16 WMMA pipeline.
//   B=128, L=1024, D_IN=64, H=512, D_OUT=64
// Pipeline (all on `stream`):
//   prep:  transpose+cvt weights to bf16 B-ready layout Bt[k][n];
//          Wc0 = W_ih0 @ W_in (fuses proj_in with layer0 input GEMM);
//          b0 = b_ih0+b_hh0, b1 = b_ih1+b_hh1
//   g1:    xw0 = u @ Wc0^T + b0            -> bufA (bf16)   [K=64]
//   rnn:   layer0 recurrence (xw0, W_hh0)  -> bufB (bf16)   [TDM prefetch of xw]
//   g2:    xw1 = y0 @ W_ih1^T + b1         -> bufA (bf16)   [K=512]
//   rnn:   layer1 recurrence (xw1, W_hh1)  -> bufB (bf16)
//   g3:    out = y1 @ W_out^T              -> d_out (f32)   [N=64]
// ---------------------------------------------------------------------------

#define BDIM   128
#define LDIM   1024
#define HDIM   512
#define DIN    64
#define DOUT   64
#define MROWS  (BDIM * LDIM)      // 131072
#define HSTRIDE 520               // LDS row stride (bf16 elems), bank-skewed, 16B-multiple

typedef __attribute__((ext_vector_type(16))) __bf16 v16bf;
typedef __attribute__((ext_vector_type(8)))  float  v8f;

union Frag {
    v16bf v;
    uint4 q[2];
    unsigned short u[16];
};

__device__ __forceinline__ unsigned short f2bf(float x) {
    unsigned int b = __float_as_uint(x);
    b += 0x7FFFu + ((b >> 16) & 1u);       // round-to-nearest-even
    return (unsigned short)(b >> 16);
}
__device__ __forceinline__ float bf2f(unsigned short h) {
    return __uint_as_float(((unsigned int)h) << 16);
}
__device__ __forceinline__ v8f wmma_bf16(const Frag& a, const Frag& b, v8f c) {
    return __builtin_amdgcn_wmma_f32_16x16x32_bf16(
        /*neg_a=*/false, a.v, /*neg_b=*/false, b.v,
        /*c_mod=*/(short)0, c, /*reuse_a=*/false, /*reuse_b=*/false);
}

// ---------------------------------------------------------------------------
// Tensor Data Mover: 2D tile load (global -> LDS).
//   tile_d1 rows of tile_d0 elements (data_size = 2 bytes, bf16),
//   row-to-row global stride d0_stride (elements).
// D# packing per CDNA5 ISA ch.8 (group0: count/lds_addr/global_addr/type=2;
// group1: data_size, tensor dims, tile dims, dim0 stride).
// ---------------------------------------------------------------------------
#if __has_builtin(__builtin_amdgcn_tensor_load_to_lds) && \
    __has_builtin(__builtin_amdgcn_s_wait_tensorcnt)
#define USE_TDM 1
#else
#define USE_TDM 0
#endif

#if USE_TDM
typedef __attribute__((ext_vector_type(4))) unsigned int v4u;
typedef __attribute__((ext_vector_type(8))) int v8i;
typedef __attribute__((ext_vector_type(4))) int v4i;

__device__ __forceinline__ void tdm_load_tile_2d_bf16(
        unsigned lds_addr, const void* gptr,
        unsigned tile_d0, unsigned tile_d1,
        unsigned tensor_d0, unsigned tensor_d1,
        unsigned long long d0_stride /* elements */) {
    unsigned long long ga = (unsigned long long)gptr;
    v4u g0;
    g0[0] = 1u;                                           // count=1 (valid), user mode
    g0[1] = lds_addr;                                     // LDS byte address
    g0[2] = (unsigned)(ga & 0xFFFFFFFFull);               // global_addr[31:0]
    g0[3] = (unsigned)((ga >> 32) & 0x01FFFFFFull)        // global_addr[56:32]
            | (2u << 30);                                 // type = 2 ("image")
    v8i g1;
    const unsigned data_size = 1;                         // 2-byte elements
    g1[0] = (int)(data_size << 16);                       // wg_mask=0, no pad/iter
    g1[1] = (int)((tensor_d0 & 0xFFFFu) << 16);           // tensor_dim0[15:0]
    g1[2] = (int)(((tensor_d0 >> 16) & 0xFFFFu)           // tensor_dim0[31:16]
            | ((tensor_d1 & 0xFFFFu) << 16));             // tensor_dim1[15:0]
    g1[3] = (int)(((tensor_d1 >> 16) & 0xFFFFu)           // tensor_dim1[31:16]
            | ((tile_d0 & 0xFFFFu) << 16));               // tile_dim0
    g1[4] = (int)(tile_d1 & 0xFFFFu);                     // tile_dim1, tile_dim2=0
    g1[5] = (int)(d0_stride & 0xFFFFFFFFull);             // dim0_stride[31:0]
    g1[6] = (int)((d0_stride >> 32) & 0xFFFFull);         // dim0_stride[47:32]
    g1[7] = 0;                                            // dim1_stride (unused, 2D tile)
    v4i gz = {0, 0, 0, 0};
#if defined(__clang_major__) && (__clang_major__ >= 23)
    v8i gz8 = {0, 0, 0, 0, 0, 0, 0, 0};
    __builtin_amdgcn_tensor_load_to_lds(g0, g1, gz, gz, gz8, 0);
#else
    __builtin_amdgcn_tensor_load_to_lds(g0, g1, gz, gz, 0);
#endif
}
#endif // USE_TDM

// ---------------------------------------------------------------------------
// Prep kernels
// ---------------------------------------------------------------------------

// dst[c*R + r] = bf16(src[r*C + c])   (B-ready transpose: Bt[k][n] = W[n][k])
__global__ void k_transpose_bf16(const float* __restrict__ src,
                                 unsigned short* __restrict__ dst,
                                 int R, int C) {
    int id = blockIdx.x * blockDim.x + threadIdx.x;
    if (id >= R * C) return;
    int r = id / C, c = id % C;
    dst[(size_t)c * R + r] = f2bf(src[(size_t)r * C + c]);
}

// Wc0[g][d] = sum_h W_ih0[g][h] * W_in[h][d];  store transposed: dst[d*512+g]
__global__ void k_fuse_in(const float* __restrict__ W_ih0,
                          const float* __restrict__ W_in,
                          unsigned short* __restrict__ dst) {
    int id = blockIdx.x * blockDim.x + threadIdx.x;   // 512*64 threads
    if (id >= HDIM * DIN) return;
    int g = id >> 6, d = id & 63;
    float s = 0.f;
    for (int h = 0; h < HDIM; ++h)
        s += W_ih0[(size_t)g * HDIM + h] * W_in[(size_t)h * DIN + d];
    dst[(size_t)d * HDIM + g] = f2bf(s);
}

__global__ void k_bias(const float* __restrict__ bi0, const float* __restrict__ bh0,
                       const float* __restrict__ bi1, const float* __restrict__ bh1,
                       float* __restrict__ b0, float* __restrict__ b1) {
    int id = blockIdx.x * blockDim.x + threadIdx.x;
    if (id >= HDIM) return;
    b0[id] = bi0[id] + bh0[id];
    b1[id] = bi1[id] + bh1[id];
}

// ---------------------------------------------------------------------------
// GEMM 1: xw0 = u(f32,[M,64]) @ Wc0^T + b0 -> bf16 [M,512]
// Block = 1 M-tile (16 rows), 8 waves, wave w -> N-tiles 4w..4w+3. K=64.
// ---------------------------------------------------------------------------
__global__ void __launch_bounds__(256)
k_gemm_in(const float* __restrict__ u, const unsigned short* __restrict__ Bt,
          const float* __restrict__ bias, unsigned short* __restrict__ out) {
    const int mt   = blockIdx.x;
    const int lane = threadIdx.x & 31;
    const int w    = threadIdx.x >> 5;
    const int half = lane >> 4;
    const int mrow = lane & 15;
    const int ncol = lane & 15;

    const float* arow = u + (size_t)(mt * 16 + mrow) * DIN;
    v8f z = {0.f,0.f,0.f,0.f,0.f,0.f,0.f,0.f};
    v8f acc[4] = {z, z, z, z};

#pragma unroll
    for (int kt = 0; kt < 2; ++kt) {
        // A fragment: f32 load + cvt to bf16 (ISA 16-bit A 16x32 layout)
        Frag fa;
        const int kb = kt * 32 + half * 8;
        union { float4 f4[4]; float f[16]; } t;
        t.f4[0] = *(const float4*)(arow + kb);
        t.f4[1] = *(const float4*)(arow + kb + 4);
        t.f4[2] = *(const float4*)(arow + kb + 16);
        t.f4[3] = *(const float4*)(arow + kb + 20);
#pragma unroll
        for (int e = 0; e < 16; ++e) fa.u[e] = f2bf(t.f[e]);

#pragma unroll
        for (int i = 0; i < 4; ++i) {
            const int n0 = (w * 4 + i) * 16;
            const unsigned short* brow = Bt + (size_t)(kt * 32) * HDIM + n0 + ncol;
            Frag fb;
#pragma unroll
            for (int e = 0; e < 16; ++e) fb.u[e] = brow[(size_t)(half * 8 + (e & 7) + (e >> 3) * 16) * HDIM];
            acc[i] = wmma_bf16(fa, fb, acc[i]);
        }
    }

#pragma unroll
    for (int i = 0; i < 4; ++i) {
        const int n = (w * 4 + i) * 16 + ncol;
        const float bv = bias[n];
#pragma unroll
        for (int j = 0; j < 8; ++j) {
            const int m = half * 8 + j;
            out[(size_t)(mt * 16 + m) * HDIM + n] = f2bf(acc[i][j] + bv);
        }
    }
}

// ---------------------------------------------------------------------------
// GEMM 2: xw1 = A(bf16,[M,512]) @ W^T + b -> bf16 [M,512].  K=512.
// ---------------------------------------------------------------------------
__global__ void __launch_bounds__(256)
k_gemm_h(const unsigned short* __restrict__ A, const unsigned short* __restrict__ Bt,
         const float* __restrict__ bias, unsigned short* __restrict__ out) {
    const int mt   = blockIdx.x;
    const int lane = threadIdx.x & 31;
    const int w    = threadIdx.x >> 5;
    const int half = lane >> 4;
    const int mrow = lane & 15;
    const int ncol = lane & 15;

    const unsigned short* arow = A + (size_t)(mt * 16 + mrow) * HDIM;
    v8f z = {0.f,0.f,0.f,0.f,0.f,0.f,0.f,0.f};
    v8f acc[4] = {z, z, z, z};

#pragma unroll 4
    for (int kt = 0; kt < 16; ++kt) {
        Frag fa;
        const int kb = kt * 32 + half * 8;
        fa.q[0] = *(const uint4*)(arow + kb);
        fa.q[1] = *(const uint4*)(arow + kb + 16);
#pragma unroll
        for (int i = 0; i < 4; ++i) {
            const int n0 = (w * 4 + i) * 16;
            const unsigned short* brow = Bt + (size_t)(kt * 32) * HDIM + n0 + ncol;
            Frag fb;
#pragma unroll
            for (int e = 0; e < 16; ++e) fb.u[e] = brow[(size_t)(half * 8 + (e & 7) + (e >> 3) * 16) * HDIM];
            acc[i] = wmma_bf16(fa, fb, acc[i]);
        }
    }

#pragma unroll
    for (int i = 0; i < 4; ++i) {
        const int n = (w * 4 + i) * 16 + ncol;
        const float bv = bias[n];
#pragma unroll
        for (int j = 0; j < 8; ++j) {
            const int m = half * 8 + j;
            out[(size_t)(mt * 16 + m) * HDIM + n] = f2bf(acc[i][j] + bv);
        }
    }
}

// ---------------------------------------------------------------------------
// GEMM 3: out = A(bf16,[M,512]) @ W_out^T -> f32 [M,64].
// Block = 2 M-tiles; wave w -> (mtile = w>>2, ntile = w&3). K=512.
// ---------------------------------------------------------------------------
__global__ void __launch_bounds__(256)
k_gemm_out(const unsigned short* __restrict__ A, const unsigned short* __restrict__ Bt,
           float* __restrict__ out) {
    const int lane = threadIdx.x & 31;
    const int w    = threadIdx.x >> 5;
    const int half = lane >> 4;
    const int mrow = lane & 15;
    const int ncol = lane & 15;
    const int mt   = blockIdx.x * 2 + (w >> 2);
    const int nt   = w & 3;

    const unsigned short* arow = A + (size_t)(mt * 16 + mrow) * HDIM;
    v8f acc = {0.f,0.f,0.f,0.f,0.f,0.f,0.f,0.f};

#pragma unroll 4
    for (int kt = 0; kt < 16; ++kt) {
        Frag fa;
        const int kb = kt * 32 + half * 8;
        fa.q[0] = *(const uint4*)(arow + kb);
        fa.q[1] = *(const uint4*)(arow + kb + 16);
        const unsigned short* brow = Bt + (size_t)(kt * 32) * DOUT + nt * 16 + ncol;
        Frag fb;
#pragma unroll
        for (int e = 0; e < 16; ++e) fb.u[e] = brow[(size_t)(half * 8 + (e & 7) + (e >> 3) * 16) * DOUT];
        acc = wmma_bf16(fa, fb, acc);
    }

#pragma unroll
    for (int j = 0; j < 8; ++j) {
        const int m = half * 8 + j;
        out[(size_t)(mt * 16 + m) * DOUT + nt * 16 + ncol] = acc[j];
    }
}

// ---------------------------------------------------------------------------
// RNN recurrence (persistent over time).
// Grid = 8 blocks (batch slices of 16 rows). 8 waves; wave w owns N cols
// [64w, 64w+64). h state (16 x 512 bf16) lives in LDS. Per step:
//   acc = h @ W_hh^T (16 bf16-WMMA K-steps x 4 N-tiles per wave)
//   h'  = tanh(acc + xw[t]);  write h' to global y and to LDS for t+1.
// W_hh^T streamed from L2 (512 KB bf16, L2-resident for all 1024 steps).
// xw[t] tile (16 x 512 bf16 = 16 KB) is double-buffered in LDS via the
// Tensor Data Mover: wave 0 issues tensor_load_to_lds for step t+1 while
// all waves compute step t; consumption gated by s_wait_tensorcnt + barrier.
// ---------------------------------------------------------------------------
__global__ void __launch_bounds__(256)
k_rnn(const unsigned short* __restrict__ xw, const unsigned short* __restrict__ Bt,
      unsigned short* __restrict__ y) {
    __shared__ __align__(16) unsigned short h_lds[16 * HSTRIDE];
    __shared__ __align__(16) unsigned short xw_stage[2][16 * HDIM];

    const int tid  = threadIdx.x;
    const int lane = tid & 31;
    const int w    = tid >> 5;
    const int half = lane >> 4;
    const int mrow = lane & 15;
    const int ncol = lane & 15;
    const int brow0 = blockIdx.x * 16;   // batch slice base

    for (int i = tid; i < 16 * HSTRIDE; i += 256) h_lds[i] = 0;

#if USE_TDM
    // Prologue: DMA the t=0 tile into stage buffer 0.
    if (w == 0) {
        tdm_load_tile_2d_bf16(
            (unsigned)(unsigned long long)&xw_stage[0][0],
            xw + ((size_t)brow0 * LDIM + 0) * HDIM,
            /*tile_d0=*/HDIM, /*tile_d1=*/16,
            /*tensor_d0=*/HDIM, /*tensor_d1=*/16,
            /*d0_stride=*/(unsigned long long)LDIM * HDIM);
    }
#endif
    __syncthreads();

    int buf = 0;
    for (int t = 0; t < LDIM; ++t) {
#if USE_TDM
        if (w == 0) __builtin_amdgcn_s_wait_tensorcnt(0);  // tile t resident
        __syncthreads();                                   // ...visible to all waves
        if (w == 0 && t + 1 < LDIM) {                      // overlap DMA of t+1
            tdm_load_tile_2d_bf16(
                (unsigned)(unsigned long long)&xw_stage[buf ^ 1][0],
                xw + ((size_t)brow0 * LDIM + (t + 1)) * HDIM,
                HDIM, 16, HDIM, 16,
                (unsigned long long)LDIM * HDIM);
        }
#else
        // Fallback: coalesced uint4 copy of tile t into the staging buffer.
        {
            uint4* dst = (uint4*)&xw_stage[buf][0];
#pragma unroll
            for (int k = 0; k < 4; ++k) {
                const int c = tid + k * 256;               // 1024 uint4 chunks
                const int m = c >> 6, off = (c & 63) * 8;
                dst[c] = *(const uint4*)(xw + ((size_t)(brow0 + m) * LDIM + t) * HDIM + off);
            }
        }
        __syncthreads();
#endif
        const unsigned short* xs = &xw_stage[buf][0];

        v8f z = {0.f,0.f,0.f,0.f,0.f,0.f,0.f,0.f};
        v8f acc[4] = {z, z, z, z};

#pragma unroll 4
        for (int kt = 0; kt < 16; ++kt) {
            Frag fa;
            const unsigned short* hrow = h_lds + mrow * HSTRIDE;
            const int kb = kt * 32 + half * 8;
            fa.q[0] = *(const uint4*)(hrow + kb);
            fa.q[1] = *(const uint4*)(hrow + kb + 16);
#pragma unroll
            for (int i = 0; i < 4; ++i) {
                const int n0 = w * 64 + i * 16;
                const unsigned short* bp = Bt + (size_t)(kt * 32) * HDIM + n0 + ncol;
                Frag fb;
#pragma unroll
                for (int e = 0; e < 16; ++e) fb.u[e] = bp[(size_t)(half * 8 + (e & 7) + (e >> 3) * 16) * HDIM];
                acc[i] = wmma_bf16(fa, fb, acc[i]);
            }
        }

        unsigned short hb[4][8];
#pragma unroll
        for (int i = 0; i < 4; ++i) {
            const int n = w * 64 + i * 16 + ncol;
#pragma unroll
            for (int j = 0; j < 8; ++j) {
                const int m = half * 8 + j;
                const float val = acc[i][j] + bf2f(xs[m * HDIM + n]);
                const float hv = tanhf(val);
                const unsigned short hx = f2bf(hv);
                hb[i][j] = hx;
                y[((size_t)(brow0 + m) * LDIM + t) * HDIM + n] = hx;
            }
        }

        __syncthreads();                          // all waves done reading old h
#pragma unroll
        for (int i = 0; i < 4; ++i) {
            const int n = w * 64 + i * 16 + ncol;
#pragma unroll
            for (int j = 0; j < 8; ++j)
                h_lds[(half * 8 + j) * HSTRIDE + n] = hb[i][j];
        }
        __syncthreads();                          // new h visible for next step
        buf ^= 1;
    }
}

// ---------------------------------------------------------------------------
// Host launcher
// ---------------------------------------------------------------------------
extern "C" void kernel_launch(void* const* d_in, const int* in_sizes, int n_in,
                              void* d_out, int out_size, void* d_ws, size_t ws_size,
                              hipStream_t stream) {
    const float* u     = (const float*)d_in[0];
    const float* W_in  = (const float*)d_in[1];
    const float* W_ih0 = (const float*)d_in[2];
    const float* W_hh0 = (const float*)d_in[3];
    const float* b_ih0 = (const float*)d_in[4];
    const float* b_hh0 = (const float*)d_in[5];
    const float* W_ih1 = (const float*)d_in[6];
    const float* W_hh1 = (const float*)d_in[7];
    const float* b_ih1 = (const float*)d_in[8];
    const float* b_hh1 = (const float*)d_in[9];
    const float* W_out = (const float*)d_in[10];

    char* ws = (char*)d_ws;
    // weight area (B-ready bf16 transposes)
    unsigned short* Wt_hh0 = (unsigned short*)(ws + 0);            // 512*512 bf16 = 512 KB
    unsigned short* Wt_hh1 = (unsigned short*)(ws + (512u << 10)); // +512 KB
    unsigned short* Wt_ih1 = (unsigned short*)(ws + (1024u << 10));// +512 KB
    unsigned short* Wt_c0  = (unsigned short*)(ws + (1536u << 10));// 64*512 bf16 = 64 KB
    unsigned short* Wt_out = (unsigned short*)(ws + (1600u << 10));// 512*64 bf16 = 64 KB
    float*          b0     = (float*)(ws + (1664u << 10));         // 2 KB
    float*          b1     = (float*)(ws + (1666u << 10));         // 2 KB
    // activation buffers: [M=131072, 512] bf16 = 128 MB each
    unsigned short* bufA = (unsigned short*)(ws + (2048ull << 10));
    unsigned short* bufB = bufA + (size_t)MROWS * HDIM;

    // ---- prep ----
    k_transpose_bf16<<<(HDIM*HDIM + 255) / 256, 256, 0, stream>>>(W_hh0, Wt_hh0, HDIM, HDIM);
    k_transpose_bf16<<<(HDIM*HDIM + 255) / 256, 256, 0, stream>>>(W_hh1, Wt_hh1, HDIM, HDIM);
    k_transpose_bf16<<<(HDIM*HDIM + 255) / 256, 256, 0, stream>>>(W_ih1, Wt_ih1, HDIM, HDIM);
    k_transpose_bf16<<<(DOUT*HDIM + 255) / 256, 256, 0, stream>>>(W_out, Wt_out, DOUT, HDIM);
    k_fuse_in<<<(HDIM*DIN + 255) / 256, 256, 0, stream>>>(W_ih0, W_in, Wt_c0);
    k_bias<<<2, 256, 0, stream>>>(b_ih0, b_hh0, b_ih1, b_hh1, b0, b1);

    // ---- pipeline ----
    k_gemm_in <<<MROWS / 16, 256, 0, stream>>>(u, Wt_c0, b0, bufA);          // xw0
    k_rnn     <<<BDIM / 16,  256, 0, stream>>>(bufA, Wt_hh0, bufB);          // layer 0
    k_gemm_h  <<<MROWS / 16, 256, 0, stream>>>(bufB, Wt_ih1, b1, bufA);      // xw1
    k_rnn     <<<BDIM / 16,  256, 0, stream>>>(bufA, Wt_hh1, bufB);          // layer 1
    k_gemm_out<<<MROWS / 32, 256, 0, stream>>>(bufB, Wt_out, (float*)d_out); // proj_out
}
